// DynamicGIN_26869315404009
// MI455X (gfx1250) — matrix operand order repaired and checked by reading, not compiled, not run on hardware
//
#include <hip/hip_runtime.h>
#include <hip/hip_bf16.h>

// ---------------------------------------------------------------------------
// Types for CDNA5 WMMA (wave32)
// ---------------------------------------------------------------------------
typedef __bf16  v16bf __attribute__((ext_vector_type(16)));
typedef float   v8f   __attribute__((ext_vector_type(8)));
typedef unsigned int v8u __attribute__((ext_vector_type(8)));
typedef unsigned int v4u __attribute__((ext_vector_type(4)));

__device__ __forceinline__ unsigned short f2bf(float f) {
    unsigned u = __float_as_uint(f);
    unsigned r = u + 0x7FFFu + ((u >> 16) & 1u);   // round-to-nearest-even
    return (unsigned short)(r >> 16);
}

// Load 16 bf16 as two 16-byte chunks -> v16bf fragment
__device__ __forceinline__ v16bf load_frag(const unsigned short* p0,
                                           const unsigned short* p1) {
    v4u a = *reinterpret_cast<const v4u*>(p0);
    v4u b = *reinterpret_cast<const v4u*>(p1);
    v8u u;
    u[0]=a[0]; u[1]=a[1]; u[2]=a[2]; u[3]=a[3];
    u[4]=b[0]; u[5]=b[1]; u[6]=b[2]; u[7]=b[3];
    return __builtin_bit_cast(v16bf, u);
}

__device__ __forceinline__ v8f wmma_bf16(v16bf a, v16bf b, v8f c) {
    return __builtin_amdgcn_wmma_f32_16x16x32_bf16(
        false, a, false, b, (short)0, c, false, false);
}

// ---------------------------------------------------------------------------
// Utility kernels
// ---------------------------------------------------------------------------
__global__ void k_zero(float* p, long long n) {
    long long i = (long long)blockIdx.x * blockDim.x + threadIdx.x;
    long long stride = (long long)gridDim.x * blockDim.x;
    for (; i < n; i += stride) p[i] = 0.0f;
}

// w (fi x fo, row-major f32) -> wt (fo x fi, row-major bf16)  [B-operand layout]
__global__ void k_convert_wt(const float* __restrict__ w,
                             unsigned short* __restrict__ wt,
                             int fi, int fo) {
    int i = blockIdx.x * blockDim.x + threadIdx.x;
    if (i >= fi * fo) return;
    int k = i / fo, n = i % fo;
    wt[(size_t)n * fi + k] = f2bf(w[i]);
}

// hin_bf16 = bf16(x + agg)
__global__ void k_prep_hin(const float* __restrict__ x,
                           const float* __restrict__ agg,
                           unsigned short* __restrict__ hin,
                           long long n) {
    long long i = (long long)blockIdx.x * blockDim.x + threadIdx.x;
    if (i >= n) return;
    hin[i] = f2bf(x[i] + agg[i]);
}

// ---------------------------------------------------------------------------
// Edge aggregation: agg[dst] += x[src]   (F compile-time, power of two)
// ---------------------------------------------------------------------------
template <int F>
__global__ void k_edge_agg(const float* __restrict__ x,
                           const int* __restrict__ src,
                           const int* __restrict__ dst,
                           float* __restrict__ agg, int E) {
    constexpr int F4 = F >> 2;
    long long tid = (long long)blockIdx.x * blockDim.x + threadIdx.x;
    if (tid >= (long long)E * F4) return;
    int e = (int)(tid >> (F == 128 ? 5 : 6));
    int c = ((int)tid & (F4 - 1)) << 2;
    int s = src[e], d = dst[e];
    float4 v = *reinterpret_cast<const float4*>(x + (size_t)s * F + c);
    float* ap = agg + (size_t)d * F + c;
    atomicAdd(ap + 0, v.x);
    atomicAdd(ap + 1, v.y);
    atomicAdd(ap + 2, v.z);
    atomicAdd(ap + 3, v.w);
}

// ---------------------------------------------------------------------------
// WMMA GEMM: C(Nrows x Nout) = A(Nrows x K, bf16) * Wt^T + bias
//   Wt stored as (Nout x K) bf16 (transposed weights).
//   Each wave: 16 rows x 64 cols. Block = 8 waves = 128 rows.
// ---------------------------------------------------------------------------
template <bool RELU, bool OUTBF16>
__global__ void k_wmma_gemm(const unsigned short* __restrict__ A,
                            const unsigned short* __restrict__ Wt,
                            const float* __restrict__ bias,
                            float* __restrict__ Cf,
                            unsigned short* __restrict__ Cb,
                            int Nrows, int K, int Nout) {
    int wave = threadIdx.x >> 5;
    int lane = threadIdx.x & 31;
    int m0 = blockIdx.y * 128 + wave * 16;
    if (m0 >= Nrows) return;
    int n0 = blockIdx.x * 64;
    int l15 = lane & 15;
    int hi = lane >> 4;

    int rowA = m0 + l15;
    if (rowA >= Nrows) rowA = Nrows - 1;      // keep EXEC full for WMMA
    const unsigned short* Arow = A + (size_t)rowA * K;

    v8f acc[4];
    acc[0] = v8f{}; acc[1] = v8f{}; acc[2] = v8f{}; acc[3] = v8f{};

    for (int k0 = 0; k0 < K; k0 += 32) {
        int ka = k0 + hi * 16;
        v16bf afrag = load_frag(Arow + ka, Arow + ka + 8);
        int kb = k0 + hi * 16;
        const unsigned short* Bbase = Wt + (size_t)(n0 + l15) * K + kb;
#pragma unroll
        for (int j = 0; j < 4; ++j) {
            const unsigned short* Bp = Bbase + (size_t)(16 * j) * K;
            v16bf bfrag = load_frag(Bp, Bp + 8);
            acc[j] = wmma_bf16(afrag, bfrag, acc[j]);
        }
    }

#pragma unroll
    for (int j = 0; j < 4; ++j) {
        int col = n0 + j * 16 + l15;
        float bv = bias[col];
#pragma unroll
        for (int r = 0; r < 8; ++r) {
            int m = m0 + r + hi * 8;
            if (m < Nrows) {
                float v = acc[j][r] + bv;
                if (RELU) v = fmaxf(v, 0.0f);
                size_t idx = (size_t)m * Nout + col;
                if (OUTBF16) Cb[idx] = f2bf(v);
                else         Cf[idx] = v;
            }
        }
    }
}

// ---------------------------------------------------------------------------
// LayerNorm + ReLU over 256-wide rows; one wave per row; optional bf16 copy.
// ---------------------------------------------------------------------------
__global__ void k_ln_relu(float* __restrict__ h,
                          unsigned short* __restrict__ hb,   // may be nullptr
                          const float* __restrict__ g,
                          const float* __restrict__ b,
                          int Nrows) {
    int row = blockIdx.x * 8 + (threadIdx.x >> 5);
    int lane = threadIdx.x & 31;
    if (row >= Nrows) return;
    float* hr = h + (size_t)row * 256;
    float v[8], s = 0.f, s2 = 0.f;
#pragma unroll
    for (int i = 0; i < 8; ++i) {
        v[i] = hr[lane + 32 * i];
        s += v[i]; s2 += v[i] * v[i];
    }
#pragma unroll
    for (int off = 16; off > 0; off >>= 1) {
        s  += __shfl_xor(s,  off, 32);
        s2 += __shfl_xor(s2, off, 32);
    }
    float mu = s * (1.0f / 256.0f);
    float var = s2 * (1.0f / 256.0f) - mu * mu;
    float inv = rsqrtf(var + 1e-5f);
#pragma unroll
    for (int i = 0; i < 8; ++i) {
        int c = lane + 32 * i;
        float o = fmaxf((v[i] - mu) * inv * g[c] + b[c], 0.0f);
        hr[c] = o;
        if (hb) hb[(size_t)row * 256 + c] = f2bf(o);
    }
}

// gate[n] = g1[n,:] . w2 + b2 ; one wave per node (128-wide rows)
__global__ void k_gate_reduce(const float* __restrict__ g1,
                              const float* __restrict__ w2,
                              const float* __restrict__ b2,
                              float* __restrict__ gate, int Nrows) {
    int row = blockIdx.x * 8 + (threadIdx.x >> 5);
    int lane = threadIdx.x & 31;
    if (row >= Nrows) return;
    const float* r = g1 + (size_t)row * 128;
    float s = 0.f;
#pragma unroll
    for (int i = 0; i < 4; ++i) {
        int c = lane + 32 * i;
        s += r[c] * w2[c];
    }
#pragma unroll
    for (int off = 16; off > 0; off >>= 1) s += __shfl_xor(s, off, 32);
    if (lane == 0) gate[row] = s + b2[0];
}

__device__ __forceinline__ int lower_bound_i(const int* a, int n, int key) {
    int lo = 0, hi = n;
    while (lo < hi) { int mid = (lo + hi) >> 1; if (a[mid] < key) lo = mid + 1; else hi = mid; }
    return lo;
}

// Per-graph softmax stats: gmax[g], den[g]
__global__ void k_graph_stats(const float* __restrict__ gate,
                              const int* __restrict__ batch, int Nn,
                              float* __restrict__ gmax, float* __restrict__ den) {
    int g = blockIdx.x, t = threadIdx.x;
    __shared__ float red[256];
    int start = lower_bound_i(batch, Nn, g);
    int end   = lower_bound_i(batch, Nn, g + 1);
    float m = -3.4e38f;
    for (int i = start + t; i < end; i += 256) m = fmaxf(m, gate[i]);
    red[t] = m; __syncthreads();
    for (int o = 128; o > 0; o >>= 1) { if (t < o) red[t] = fmaxf(red[t], red[t + o]); __syncthreads(); }
    float M = red[0]; __syncthreads();
    float s = 0.f;
    for (int i = start + t; i < end; i += 256) s += __expf(gate[i] - M);
    red[t] = s; __syncthreads();
    for (int o = 128; o > 0; o >>= 1) { if (t < o) red[t] += red[t + o]; __syncthreads(); }
    if (t == 0) { gmax[g] = M; den[g] = (end > start) ? red[0] : 1.0f; }
}

// pooled[g,:] = sum_n h[n,:] * alpha[n]; deterministic per-graph reduction
__global__ void k_pool(const float* __restrict__ h,
                       const float* __restrict__ gate,
                       const int* __restrict__ batch, int Nn,
                       const float* __restrict__ gmax,
                       const float* __restrict__ den,
                       float* __restrict__ pooled) {
    int g = blockIdx.x, t = threadIdx.x;      // 256 threads <-> 256 cols
    __shared__ float al[256];
    int start = lower_bound_i(batch, Nn, g);
    int end   = lower_bound_i(batch, Nn, g + 1);
    float M = gmax[g], inv = 1.0f / den[g];
    float acc = 0.f;
    for (int base = start; base < end; base += 256) {
        int i = base + t;
        al[t] = (i < end) ? __expf(gate[i] - M) * inv : 0.0f;
        __syncthreads();
        int cnt = min(256, end - base);
        for (int j = 0; j < cnt; ++j)
            acc += h[(size_t)(base + j) * 256 + t] * al[j];
        __syncthreads();
    }
    pooled[(size_t)g * 256 + t] = acc;
}

// out[g,0:2] = ReLU(pooled[g] @ cls_w1 + b1) @ cls_w2 + b2
__global__ void k_classify(const float* __restrict__ pooled,
                           const float* __restrict__ w1, const float* __restrict__ b1,
                           const float* __restrict__ w2, const float* __restrict__ b2,
                           float* __restrict__ out) {
    int g = blockIdx.x, t = threadIdx.x;      // 128 threads
    __shared__ float pr[256];
    __shared__ float p[128];
    pr[t] = pooled[(size_t)g * 256 + t];
    pr[t + 128] = pooled[(size_t)g * 256 + t + 128];
    __syncthreads();
    float s = b1[t];
    for (int k = 0; k < 256; ++k) s += pr[k] * w1[k * 128 + t];
    p[t] = fmaxf(s, 0.0f);
    __syncthreads();
    if (t < 2) {
        float o = b2[t];
        for (int k = 0; k < 128; ++k) o += p[k] * w2[k * 2 + t];
        out[(size_t)g * 2 + t] = o;
    }
}

// ---------------------------------------------------------------------------
// Launch
// ---------------------------------------------------------------------------
extern "C" void kernel_launch(void* const* d_in, const int* in_sizes, int n_in,
                              void* d_out, int out_size, void* d_ws, size_t ws_size,
                              hipStream_t stream) {
    const float* x       = (const float*)d_in[0];
    const int*   eidx    = (const int*)d_in[1];
    const int*   batch   = (const int*)d_in[2];
    const float* w1_0    = (const float*)d_in[4];
    const float* b1_0    = (const float*)d_in[5];
    const float* w2_0    = (const float*)d_in[6];
    const float* b2_0    = (const float*)d_in[7];
    const float* ln_g0   = (const float*)d_in[8];
    const float* ln_b0   = (const float*)d_in[9];
    const float* w1_1    = (const float*)d_in[10];
    const float* b1_1    = (const float*)d_in[11];
    const float* w2_1    = (const float*)d_in[12];
    const float* b2_1    = (const float*)d_in[13];
    const float* ln_g1   = (const float*)d_in[14];
    const float* ln_b1   = (const float*)d_in[15];
    const float* gate_w1 = (const float*)d_in[16];
    const float* gate_b1 = (const float*)d_in[17];
    const float* gate_w2 = (const float*)d_in[18];
    const float* gate_b2 = (const float*)d_in[19];
    const float* cls_w1  = (const float*)d_in[20];
    const float* cls_b1  = (const float*)d_in[21];
    const float* cls_w2  = (const float*)d_in[22];
    const float* cls_b2  = (const float*)d_in[23];
    float* out = (float*)d_out;

    const int N = in_sizes[0] / 128;     // 100000
    const int E = in_sizes[1] / 2;       // 1600000
    const int G = out_size / 2;          // 128
    const int* src = eidx;
    const int* dst = eidx + E;

    // ---- workspace carve-up ----
    char* ws = (char*)d_ws;
    size_t off = 0;
    auto carve = [&](size_t bytes) { char* p = ws + off; off += (bytes + 255) & ~(size_t)255; return p; };
    float*          agg    = (float*)         carve((size_t)N * 256 * 4);
    float*          hbuf   = (float*)         carve((size_t)N * 256 * 4);
    unsigned short* bfA    = (unsigned short*)carve((size_t)N * 256 * 2);
    unsigned short* bfB    = (unsigned short*)carve((size_t)N * 256 * 2);
    float*          gate   = (float*)         carve((size_t)N * 4);
    unsigned short* w1_0t  = (unsigned short*)carve(256 * 128 * 2);
    unsigned short* w2_0t  = (unsigned short*)carve(256 * 256 * 2);
    unsigned short* w1_1t  = (unsigned short*)carve(256 * 256 * 2);
    unsigned short* w2_1t  = (unsigned short*)carve(256 * 256 * 2);
    unsigned short* gw1t   = (unsigned short*)carve(128 * 256 * 2);
    float*          gmax   = (float*)         carve((size_t)G * 4);
    float*          den    = (float*)         carve((size_t)G * 4);
    float*          pooled = (float*)         carve((size_t)G * 256 * 4);
    (void)ws_size; (void)n_in;

    // ---- weight conversion (f32 row-major -> bf16 transposed) ----
    k_convert_wt<<<(128 * 256 + 255) / 256, 256, 0, stream>>>(w1_0,    w1_0t, 128, 256);
    k_convert_wt<<<(256 * 256 + 255) / 256, 256, 0, stream>>>(w2_0,    w2_0t, 256, 256);
    k_convert_wt<<<(256 * 256 + 255) / 256, 256, 0, stream>>>(w1_1,    w1_1t, 256, 256);
    k_convert_wt<<<(256 * 256 + 255) / 256, 256, 0, stream>>>(w2_1,    w2_1t, 256, 256);
    k_convert_wt<<<(256 * 128 + 255) / 256, 256, 0, stream>>>(gate_w1, gw1t,  256, 128);

    const dim3 gemm_grid4(4, (N + 127) / 128);   // Nout=256
    const dim3 gemm_grid2(2, (N + 127) / 128);   // Nout=128
    const int  ln_blocks = (N + 7) / 8;

    // ================= Layer 0 (F=128) =================
    k_zero<<<2048, 256, 0, stream>>>(agg, (long long)N * 128);
    {
        long long tot = (long long)E * 32;
        k_edge_agg<128><<<(unsigned)((tot + 255) / 256), 256, 0, stream>>>(x, src, dst, agg, E);
    }
    {
        long long tot = (long long)N * 128;
        k_prep_hin<<<(unsigned)((tot + 255) / 256), 256, 0, stream>>>(x, agg, bfA, tot);
    }
    k_wmma_gemm<true,  true ><<<gemm_grid4, 256, 0, stream>>>(bfA, w1_0t, b1_0, nullptr, bfB, N, 128, 256);
    k_wmma_gemm<false, false><<<gemm_grid4, 256, 0, stream>>>(bfB, w2_0t, b2_0, hbuf, nullptr, N, 256, 256);
    k_ln_relu<<<ln_blocks, 256, 0, stream>>>(hbuf, nullptr, ln_g0, ln_b0, N);

    // ================= Layer 1 (F=256) =================
    k_zero<<<2048, 256, 0, stream>>>(agg, (long long)N * 256);
    {
        long long tot = (long long)E * 64;
        k_edge_agg<256><<<(unsigned)((tot + 255) / 256), 256, 0, stream>>>(hbuf, src, dst, agg, E);
    }
    {
        long long tot = (long long)N * 256;
        k_prep_hin<<<(unsigned)((tot + 255) / 256), 256, 0, stream>>>(hbuf, agg, bfA, tot);
    }
    k_wmma_gemm<true,  true ><<<gemm_grid4, 256, 0, stream>>>(bfA, w1_1t, b1_1, nullptr, bfB, N, 256, 256);
    k_wmma_gemm<false, false><<<gemm_grid4, 256, 0, stream>>>(bfB, w2_1t, b2_1, hbuf, nullptr, N, 256, 256);
    k_ln_relu<<<ln_blocks, 256, 0, stream>>>(hbuf, bfA, ln_g1, ln_b1, N);   // also emit bf16 h

    // ================= Gate + attention pooling =================
    float* g1 = (float*)bfB;   // reuse: N*256*2 bytes == N*128 floats
    k_wmma_gemm<true, false><<<gemm_grid2, 256, 0, stream>>>(bfA, gw1t, gate_b1, g1, nullptr, N, 256, 128);
    k_gate_reduce<<<ln_blocks, 256, 0, stream>>>(g1, gate_w2, gate_b2, gate, N);
    k_graph_stats<<<G, 256, 0, stream>>>(gate, batch, N, gmax, den);
    k_pool<<<G, 256, 0, stream>>>(hbuf, gate, batch, N, gmax, den, pooled);

    // ================= Classifier =================
    k_classify<<<G, 128, 0, stream>>>(pooled, cls_w1, cls_b1, cls_w2, cls_b2, out);
}
